// ChatModel_2671469658350
// MI455X (gfx1250) — matrix operand order repaired and checked by reading, not compiled, not run on hardware
//
#include <hip/hip_runtime.h>
#include <hip/hip_bf16.h>
#include <math.h>

// ---------------- CDNA5 WMMA types ----------------
typedef __attribute__((ext_vector_type(16))) __bf16 v16bf;
typedef __attribute__((ext_vector_type(8)))  float  v8f;
typedef __attribute__((ext_vector_type(4)))  int    v4i;

union Frag { uint4 u[2]; v16bf v; };

__device__ __forceinline__ unsigned short f32_to_bf16_rne(float f) {
    unsigned int u = __float_as_uint(f);
    u += 0x7FFFu + ((u >> 16) & 1u);
    return (unsigned short)(u >> 16);
}

__device__ __forceinline__ v8f bwmma(const Frag& a, const Frag& b, v8f c) {
    return __builtin_amdgcn_wmma_f32_16x16x32_bf16(false, a.v, false, b.v,
                                                   (short)0, c, false, false);
}

// ---- async global->LDS staging (CDNA5 ASYNCcnt path), guarded ----
// Diagnostic-verified param type: (v4i addrspace(1)*, v4i addrspace(3)*, imm, imm)
#if __has_builtin(__builtin_amdgcn_global_load_async_to_lds_b128)
#define HAVE_ASYNC_LDS 1
#define ASYNC_B128(g, l)                                                      \
    __builtin_amdgcn_global_load_async_to_lds_b128(                           \
        (__attribute__((address_space(1))) v4i*)(g),                          \
        (__attribute__((address_space(3))) v4i*)(l), 0, 0)
#else
#define HAVE_ASYNC_LDS 0
#endif

#if __has_builtin(__builtin_amdgcn_s_wait_asynccnt)
#define WAIT_ASYNC() __builtin_amdgcn_s_wait_asynccnt(0)
#else
#define WAIT_ASYNC() asm volatile("s_wait_asynccnt 0x0" ::: "memory")
#endif

// ---- intra-16-lane butterfly reductions via v_permlane16_b32 (VALU, no LDS) ----
#if __has_builtin(__builtin_amdgcn_permlane16)
__device__ __forceinline__ float permx1(float x) {
    unsigned u = __float_as_uint(x);
    return __uint_as_float(__builtin_amdgcn_permlane16(u, u, 0x67452301u, 0xEFCDAB89u, false, false));
}
__device__ __forceinline__ float permx2(float x) {
    unsigned u = __float_as_uint(x);
    return __uint_as_float(__builtin_amdgcn_permlane16(u, u, 0x54761032u, 0xDCFE98BAu, false, false));
}
__device__ __forceinline__ float permx4(float x) {
    unsigned u = __float_as_uint(x);
    return __uint_as_float(__builtin_amdgcn_permlane16(u, u, 0x32107654u, 0xBA98FEDCu, false, false));
}
__device__ __forceinline__ float permx8(float x) {
    unsigned u = __float_as_uint(x);
    return __uint_as_float(__builtin_amdgcn_permlane16(u, u, 0xFEDCBA98u, 0x76543210u, false, false));
}
#else
__device__ __forceinline__ float permx1(float x) { return __shfl_xor(x, 1); }
__device__ __forceinline__ float permx2(float x) { return __shfl_xor(x, 2); }
__device__ __forceinline__ float permx4(float x) { return __shfl_xor(x, 4); }
__device__ __forceinline__ float permx8(float x) { return __shfl_xor(x, 8); }
#endif

__device__ __forceinline__ float red_max16(float x) {
    x = fmaxf(x, permx1(x));
    x = fmaxf(x, permx2(x));
    x = fmaxf(x, permx4(x));
    x = fmaxf(x, permx8(x));
    return x;
}
__device__ __forceinline__ float red_sum16(float x) {
    x += permx1(x);
    x += permx2(x);
    x += permx4(x);
    x += permx8(x);
    return x;
}

// ---------------- elementwise / prep kernels ----------------
__global__ void transpose_cast_kernel(const float* __restrict__ in,
                                      unsigned short* __restrict__ out, int K, int N) {
    __shared__ float tile[32][33];
    int n0 = blockIdx.x * 32;
    int k0 = blockIdx.y * 32;
    int tx = threadIdx.x;
    int ty = threadIdx.y;
    for (int i = ty; i < 32; i += 8)
        tile[i][tx] = in[(size_t)(k0 + i) * N + n0 + tx];
    __syncthreads();
    for (int i = ty; i < 32; i += 8)
        out[(size_t)(n0 + i) * K + k0 + tx] = f32_to_bf16_rne(tile[tx][i]);
}

// Vt[(b*H+h)*64 + dh][s] = Vbf[(b*S+s)*D + h*64 + dh]   (bf16 -> bf16)
__global__ void vtranspose_kernel(const unsigned short* __restrict__ Vbf,
                                  unsigned short* __restrict__ Vt,
                                  int S, int H, int D) {
    __shared__ unsigned short tile[32][33];
    int s0  = blockIdx.x * 32;
    int y   = blockIdx.y;
    int dhb = (y & 1) * 32;
    int bh  = y >> 1;
    int b = bh / H, hh = bh % H;
    int tx = threadIdx.x, ty = threadIdx.y;
    for (int i = ty; i < 32; i += 8)
        tile[i][tx] = Vbf[(size_t)(b * S + s0 + i) * D + hh * 64 + dhb + tx];
    __syncthreads();
    for (int i = ty; i < 32; i += 8)
        Vt[(size_t)(bh * 64 + dhb + i) * S + s0 + tx] = tile[tx][i];
}

__global__ void gelu_cast_kernel(const float* __restrict__ in,
                                 unsigned short* __restrict__ out, int n) {
    for (int i = blockIdx.x * 256 + threadIdx.x; i < n; i += gridDim.x * 256) {
        float x = in[i];
        float g = 0.5f * x * (1.0f + erff(x * 0.70710678118654752f));
        out[i] = f32_to_bf16_rne(g);
    }
}

__global__ void add1_kernel(float* __restrict__ h, const float* __restrict__ a, int n) {
    for (int i = blockIdx.x * 256 + threadIdx.x; i < n; i += gridDim.x * 256)
        h[i] += a[i];
}

__global__ void zero_kernel(float* __restrict__ p, int n) {
    for (int i = blockIdx.x * 256 + threadIdx.x; i < n; i += gridDim.x * 256)
        p[i] = 0.0f;
}

__global__ void embed_kernel(const int* __restrict__ x, const float* __restrict__ tok,
                             const float* __restrict__ pos, float* __restrict__ h,
                             int D, int S) {
    int row = blockIdx.x;
    int t = x[row];
    int spos = row % S;
    for (int c = threadIdx.x; c < D; c += 256)
        h[(size_t)row * D + c] = tok[(size_t)t * D + c] + pos[(size_t)spos * D + c];
}

__global__ void layernorm_kernel(const float* __restrict__ in, const float* __restrict__ g,
                                 const float* __restrict__ b, unsigned short* __restrict__ obf,
                                 float* __restrict__ of32, int D) {
    __shared__ float r1[256];
    __shared__ float r2[256];
    int row = blockIdx.x, tid = threadIdx.x;
    const float* p = in + (size_t)row * D;
    float s = 0.f, s2 = 0.f;
    for (int c = tid; c < D; c += 256) { float v = p[c]; s += v; s2 += v * v; }
    r1[tid] = s; r2[tid] = s2; __syncthreads();
    for (int st = 128; st > 0; st >>= 1) {
        if (tid < st) { r1[tid] += r1[tid + st]; r2[tid] += r2[tid + st]; }
        __syncthreads();
    }
    float mu  = r1[0] / (float)D;
    float var = r2[0] / (float)D - mu * mu;
    float inv = rsqrtf(var + 1e-5f);
    for (int c = tid; c < D; c += 256) {
        float v = (p[c] - mu) * inv * g[c] + b[c];
        obf[(size_t)row * D + c] = f32_to_bf16_rne(v);
        if (of32) of32[(size_t)row * D + c] = v;
    }
}

// ---------------- WMMA bf16 GEMM ----------------
// OUT_MODE: 0 = store f32, 1 = store bf16, 2 = accumulate into f32 (C += A@B + bias)
#define TM 128
#define TN 128
#define TK 64
#define LDSS 72

template <bool HAS_BIAS, int OUT_MODE>
__global__ __launch_bounds__(256)
void gemm_bf16_wmma_kernel(const unsigned short* __restrict__ A, int lda,
                           const unsigned short* __restrict__ BT, int ldb,
                           void* __restrict__ Cout, int ldc,
                           const float* __restrict__ bias,
                           int M, int N, int K) {
    __shared__ alignas(16) unsigned short ldsA[2][TM * LDSS];
    __shared__ alignas(16) unsigned short ldsB[2][TN * LDSS];

    const int tid  = threadIdx.x;
    const int lane = tid & 31;
    const int w    = tid >> 5;
    const int wm   = w >> 2;
    const int wn   = w & 3;
    const int m0   = blockIdx.y * TM;
    const int n0   = blockIdx.x * TN;
    const int ln16 = lane & 15;
    const int g    = lane >> 4;

    const int srow  = tid >> 1;
    const int shalf = (tid & 1) * 32;

    const unsigned short* gA = A  + (size_t)(m0 + srow) * lda + shalf;
    const unsigned short* gB = BT + (size_t)(n0 + srow) * ldb + shalf;
    unsigned short* lA0 = ldsA[0] + srow * LDSS + shalf;
    unsigned short* lB0 = ldsB[0] + srow * LDSS + shalf;
    unsigned short* lA1 = ldsA[1] + srow * LDSS + shalf;
    unsigned short* lB1 = ldsB[1] + srow * LDSS + shalf;

    v8f acc[4][2];
#pragma unroll
    for (int i = 0; i < 4; ++i)
#pragma unroll
        for (int j = 0; j < 2; ++j) acc[i][j] = (v8f){};

#if HAVE_ASYNC_LDS
    // ---- preload tile 0 via async global->LDS (ASYNCcnt) ----
    ASYNC_B128(gA,      lA0);      ASYNC_B128(gA + 8,  lA0 + 8);
    ASYNC_B128(gA + 16, lA0 + 16); ASYNC_B128(gA + 24, lA0 + 24);
    ASYNC_B128(gB,      lB0);      ASYNC_B128(gB + 8,  lB0 + 8);
    ASYNC_B128(gB + 16, lB0 + 16); ASYNC_B128(gB + 24, lB0 + 24);
    WAIT_ASYNC();
    __syncthreads();
#else
    {
        const uint4* sa = (const uint4*)gA;
        uint4* da = (uint4*)lA0;
        da[0] = sa[0]; da[1] = sa[1]; da[2] = sa[2]; da[3] = sa[3];
        const uint4* sb = (const uint4*)gB;
        uint4* db = (uint4*)lB0;
        db[0] = sb[0]; db[1] = sb[1]; db[2] = sb[2]; db[3] = sb[3];
    }
    __syncthreads();
#endif

    int cur = 0;
    for (int kb = 0; kb < K; kb += TK) {
        const bool has_next = (kb + TK) < K;
#if HAVE_ASYNC_LDS
        if (has_next) {
            const unsigned short* nA = gA + kb + TK;
            const unsigned short* nB = gB + kb + TK;
            unsigned short* dA = cur ? lA0 : lA1;
            unsigned short* dB = cur ? lB0 : lB1;
            ASYNC_B128(nA,      dA);      ASYNC_B128(nA + 8,  dA + 8);
            ASYNC_B128(nA + 16, dA + 16); ASYNC_B128(nA + 24, dA + 24);
            ASYNC_B128(nB,      dB);      ASYNC_B128(nB + 8,  dB + 8);
            ASYNC_B128(nB + 16, dB + 16); ASYNC_B128(nB + 24, dB + 24);
        }
#else
        uint4 pa0, pa1, pa2, pa3, pb0, pb1, pb2, pb3;
        if (has_next) {
            const uint4* sa = (const uint4*)(gA + kb + TK);
            pa0 = sa[0]; pa1 = sa[1]; pa2 = sa[2]; pa3 = sa[3];
            const uint4* sb = (const uint4*)(gB + kb + TK);
            pb0 = sb[0]; pb1 = sb[1]; pb2 = sb[2]; pb3 = sb[3];
        }
#endif

        const unsigned short* bufA = ldsA[cur];
        const unsigned short* bufB = ldsB[cur];
#pragma unroll
        for (int ks = 0; ks < TK; ks += 32) {
            Frag af[4], bf[2];
#pragma unroll
            for (int fm = 0; fm < 4; ++fm) {
                const unsigned short* p = bufA + (wm * 64 + fm * 16 + ln16) * LDSS + ks;
                af[fm].u[0] = *(const uint4*)(p + 8 * g);
                af[fm].u[1] = *(const uint4*)(p + 16 + 8 * g);
            }
#pragma unroll
            for (int fn = 0; fn < 2; ++fn) {
                const unsigned short* p = bufB + (wn * 32 + fn * 16 + ln16) * LDSS + ks;
                bf[fn].u[0] = *(const uint4*)(p + 8 * g);
                bf[fn].u[1] = *(const uint4*)(p + 16 + 8 * g);
            }
#pragma unroll
            for (int fm = 0; fm < 4; ++fm)
#pragma unroll
                for (int fn = 0; fn < 2; ++fn)
                    acc[fm][fn] = bwmma(af[fm], bf[fn], acc[fm][fn]);
        }

#if HAVE_ASYNC_LDS
        if (has_next) WAIT_ASYNC();
        __syncthreads();
#else
        if (has_next) {
            uint4* da = (uint4*)(cur ? lA0 : lA1);
            da[0] = pa0; da[1] = pa1; da[2] = pa2; da[3] = pa3;
            uint4* db = (uint4*)(cur ? lB0 : lB1);
            db[0] = pb0; db[1] = pb1; db[2] = pb2; db[3] = pb3;
        }
        __syncthreads();
#endif
        cur ^= 1;
    }

#pragma unroll
    for (int fm = 0; fm < 4; ++fm)
#pragma unroll
        for (int fn = 0; fn < 2; ++fn) {
            int col = n0 + wn * 32 + fn * 16 + ln16;
            float bv = HAS_BIAS ? bias[col] : 0.0f;
#pragma unroll
            for (int v = 0; v < 8; ++v) {
                int row = m0 + wm * 64 + fm * 16 + g * 8 + v;
                float val = acc[fm][fn][v] + bv;
                size_t idx = (size_t)row * ldc + col;
                if (OUT_MODE == 1)
                    ((unsigned short*)Cout)[idx] = f32_to_bf16_rne(val);
                else if (OUT_MODE == 2)
                    ((float*)Cout)[idx] += val;
                else
                    ((float*)Cout)[idx] = val;
            }
        }
}

// ---------------- WMMA flash attention (causal, online softmax) ----------------
__global__ __launch_bounds__(128)
void flash_attn_kernel(const unsigned short* __restrict__ Qbf,
                       const unsigned short* __restrict__ Kbf,
                       const unsigned short* __restrict__ Vt,
                       unsigned short* __restrict__ Obf,
                       int S, int H, int D) {
    __shared__ alignas(16) unsigned short ldsP[4][16 * 40];
    const int tid  = threadIdx.x;
    const int lane = tid & 31;
    const int w    = tid >> 5;
    const int ln16 = lane & 15;
    const int g    = lane >> 4;
    const int bh   = blockIdx.y;
    const int b    = bh / H, hh = bh % H;
    const int sq0  = blockIdx.x * 64 + w * 16;

    const unsigned qoff = (unsigned)(b * S + sq0 + ln16) * (unsigned)D + (unsigned)(hh * 64);
    Frag qf[2];
    qf[0].u[0] = *(const uint4*)(Qbf + qoff + 8 * g);
    qf[0].u[1] = *(const uint4*)(Qbf + qoff + 16 + 8 * g);
    qf[1].u[0] = *(const uint4*)(Qbf + qoff + 32 + 8 * g);
    qf[1].u[1] = *(const uint4*)(Qbf + qoff + 32 + 16 + 8 * g);

    v8f accO[4];
#pragma unroll
    for (int j = 0; j < 4; ++j) accO[j] = (v8f){};
    float m8[8], l8[8];
#pragma unroll
    for (int v = 0; v < 8; ++v) { m8[v] = -3.0e38f; l8[v] = 0.f; }

    unsigned short* myP = ldsP[w];
    const unsigned vtrow0 = (unsigned)bh * 64u;
    const int kend = sq0 + 16;

    for (int kb = 0; kb < kend; kb += 32) {
        v8f sc[2];
#pragma unroll
        for (int blk = 0; blk < 2; ++blk) {
            int sk = kb + blk * 16 + ln16;
            int skc = sk < S ? sk : (S - 1);
            const unsigned koff = (unsigned)(b * S + skc) * (unsigned)D + (unsigned)(hh * 64);
            Frag kf0, kf1;
            kf0.u[0] = *(const uint4*)(Kbf + koff + 8 * g);
            kf0.u[1] = *(const uint4*)(Kbf + koff + 16 + 8 * g);
            kf1.u[0] = *(const uint4*)(Kbf + koff + 32 + 8 * g);
            kf1.u[1] = *(const uint4*)(Kbf + koff + 32 + 16 + 8 * g);
            v8f z = (v8f){};
            z = bwmma(qf[0], kf0, z);
            z = bwmma(qf[1], kf1, z);
            sc[blk] = z;
        }

#pragma unroll
        for (int v = 0; v < 8; ++v) {
            int row = sq0 + v + 8 * g;
            float x0 = (kb + ln16)      <= row ? sc[0][v] * 0.125f : -3.0e38f;
            float x1 = (kb + 16 + ln16) <= row ? sc[1][v] * 0.125f : -3.0e38f;
            float mx = red_max16(fmaxf(x0, x1));
            float mnew = fmaxf(m8[v], mx);
            float p0 = __expf(x0 - mnew);
            float p1 = __expf(x1 - mnew);
            float rs = red_sum16(p0 + p1);
            float corr = __expf(m8[v] - mnew);
            l8[v] = l8[v] * corr + rs;
            m8[v] = mnew;
#pragma unroll
            for (int j = 0; j < 4; ++j) accO[j][v] *= corr;
            myP[(v + 8 * g) * 40 + ln16]      = f32_to_bf16_rne(p0);
            myP[(v + 8 * g) * 40 + 16 + ln16] = f32_to_bf16_rne(p1);
        }

        Frag pf;
        const unsigned short* pp = myP + ln16 * 40;
        pf.u[0] = *(const uint4*)(pp + 8 * g);
        pf.u[1] = *(const uint4*)(pp + 16 + 8 * g);
#pragma unroll
        for (int j = 0; j < 4; ++j) {
            const unsigned short* vp = Vt + (vtrow0 + j * 16 + ln16) * (unsigned)S + kb;
            Frag vf;
            vf.u[0] = *(const uint4*)(vp + 8 * g);
            vf.u[1] = *(const uint4*)(vp + 16 + 8 * g);
            accO[j] = bwmma(pf, vf, accO[j]);
        }
    }

#pragma unroll
    for (int j = 0; j < 4; ++j)
#pragma unroll
        for (int v = 0; v < 8; ++v) {
            unsigned ooff = (unsigned)(b * S + sq0 + v + 8 * g) * (unsigned)D
                          + (unsigned)(hh * 64 + j * 16 + ln16);
            Obf[ooff] = f32_to_bf16_rne(accO[j][v] / l8[v]);
        }
}

// ---------------- cross-entropy loss ----------------
__global__ void loss_kernel(const float* __restrict__ logits, const int* __restrict__ tgt,
                            float* __restrict__ loss, int V, float invN) {
    __shared__ float red[256];
    int row = blockIdx.x, tid = threadIdx.x;
    const float* p = logits + (size_t)row * V;
    float mx = -3.4e38f;
    for (int c = tid; c < V; c += 256) mx = fmaxf(mx, p[c]);
    red[tid] = mx; __syncthreads();
    for (int s = 128; s > 0; s >>= 1) {
        if (tid < s) red[tid] = fmaxf(red[tid], red[tid + s]);
        __syncthreads();
    }
    mx = red[0]; __syncthreads();
    float sm = 0.f;
    for (int c = tid; c < V; c += 256) sm += __expf(p[c] - mx);
    red[tid] = sm; __syncthreads();
    for (int s = 128; s > 0; s >>= 1) {
        if (tid < s) red[tid] += red[tid + s];
        __syncthreads();
    }
    if (tid == 0) {
        float lp = p[tgt[row]] - mx - logf(red[0]);
        atomicAdd(loss, -lp * invN);
    }
}

// ---------------- host orchestration ----------------
extern "C" void kernel_launch(void* const* d_in, const int* in_sizes, int n_in,
                              void* d_out, int out_size, void* d_ws, size_t ws_size,
                              hipStream_t stream) {
    (void)in_sizes; (void)n_in; (void)out_size; (void)ws_size;
    const int V = 32000, D = 1024, L = 8, H = 16, S = 1024, B = 2;
    const int DFF = 4096;
    const int M = B * S;

    const int*   x       = (const int*)  d_in[0];
    const int*   targets = (const int*)  d_in[1];
    const float* tok_emb = (const float*)d_in[2];
    const float* pos_emb = (const float*)d_in[3];
    const float* Wq      = (const float*)d_in[4];
    const float* Wk      = (const float*)d_in[5];
    const float* Wv      = (const float*)d_in[6];
    const float* Wo      = (const float*)d_in[7];
    const float* bo      = (const float*)d_in[8];
    const float* ln1_g   = (const float*)d_in[9];
    const float* ln1_b   = (const float*)d_in[10];
    const float* ln2_g   = (const float*)d_in[11];
    const float* ln2_b   = (const float*)d_in[12];
    const float* W1      = (const float*)d_in[13];
    const float* b1      = (const float*)d_in[14];
    const float* W2      = (const float*)d_in[15];
    const float* b2      = (const float*)d_in[16];
    const float* lnf_g   = (const float*)d_in[17];
    const float* lnf_b   = (const float*)d_in[18];
    const float* W_out   = (const float*)d_in[19];
    const float* b_out   = (const float*)d_in[20];

    char* ws = (char*)d_ws;
    size_t off = 0;
    auto alloc = [&](size_t bytes) -> void* {
        void* p = ws + off;
        off = (off + bytes + 255) & ~(size_t)255;
        return p;
    };

    unsigned short* wq_t   = (unsigned short*)alloc((size_t)L * D * D * 2);
    unsigned short* wk_t   = (unsigned short*)alloc((size_t)L * D * D * 2);
    unsigned short* wv_t   = (unsigned short*)alloc((size_t)L * D * D * 2);
    unsigned short* wo_t   = (unsigned short*)alloc((size_t)L * D * D * 2);
    unsigned short* w1_t   = (unsigned short*)alloc((size_t)L * D * DFF * 2);
    unsigned short* w2_t   = (unsigned short*)alloc((size_t)L * DFF * D * 2);
    unsigned short* wout_t = (unsigned short*)alloc((size_t)D * V * 2);

    float*          h    = (float*)alloc((size_t)M * D * 4);
    unsigned short* abf  = (unsigned short*)alloc((size_t)M * D * 2);
    unsigned short* Qbf  = (unsigned short*)alloc((size_t)M * D * 2);
    unsigned short* Kbf  = (unsigned short*)alloc((size_t)M * D * 2);
    unsigned short* Vbf  = (unsigned short*)alloc((size_t)M * D * 2);
    unsigned short* Vt   = (unsigned short*)alloc((size_t)B * H * 64 * S * 2 + 256);
    float*          Ff   = (float*)alloc((size_t)M * D * 4);
    float*          T1   = (float*)alloc((size_t)M * DFF * 4);
    unsigned short* t1bf = (unsigned short*)alloc((size_t)M * DFF * 2);

    float* logits = (float*)d_out;
    float* lossp  = logits + (size_t)M * V;

    auto tcvt = [&](const float* src, unsigned short* dst, int K, int N) {
        dim3 grid(N / 32, K / 32), blk(32, 8);
        transpose_cast_kernel<<<grid, blk, 0, stream>>>(src, dst, K, N);
    };
    for (int l = 0; l < L; ++l) {
        tcvt(Wq + (size_t)l * D * D,   wq_t + (size_t)l * D * D,   D, D);
        tcvt(Wk + (size_t)l * D * D,   wk_t + (size_t)l * D * D,   D, D);
        tcvt(Wv + (size_t)l * D * D,   wv_t + (size_t)l * D * D,   D, D);
        tcvt(Wo + (size_t)l * D * D,   wo_t + (size_t)l * D * D,   D, D);
        tcvt(W1 + (size_t)l * D * DFF, w1_t + (size_t)l * D * DFF, D, DFF);
        tcvt(W2 + (size_t)l * DFF * D, w2_t + (size_t)l * DFF * D, DFF, D);
    }
    tcvt(W_out, wout_t, D, V);

    embed_kernel<<<M, 256, 0, stream>>>(x, tok_emb, pos_emb, h, D, S);

    const dim3 gemmBlk(256);
    const int nMD = M * D;
    const int addBlocks = (nMD + 255) / 256;

    for (int l = 0; l < L; ++l) {
        layernorm_kernel<<<M, 256, 0, stream>>>(h, ln1_g + (size_t)l * D, ln1_b + (size_t)l * D,
                                                abf, (float*)nullptr, D);
        dim3 gDD(D / TN, M / TM);
        gemm_bf16_wmma_kernel<false, 1><<<gDD, gemmBlk, 0, stream>>>(
            abf, D, wq_t + (size_t)l * D * D, D, Qbf, D, nullptr, M, D, D);
        gemm_bf16_wmma_kernel<false, 1><<<gDD, gemmBlk, 0, stream>>>(
            abf, D, wk_t + (size_t)l * D * D, D, Kbf, D, nullptr, M, D, D);
        gemm_bf16_wmma_kernel<false, 1><<<gDD, gemmBlk, 0, stream>>>(
            abf, D, wv_t + (size_t)l * D * D, D, Vbf, D, nullptr, M, D, D);
        vtranspose_kernel<<<dim3(S / 32, B * H * 2), dim3(32, 8), 0, stream>>>(Vbf, Vt, S, H, D);
        flash_attn_kernel<<<dim3(S / 64, B * H), 128, 0, stream>>>(Qbf, Kbf, Vt, abf, S, H, D);
        gemm_bf16_wmma_kernel<true, 2><<<gDD, gemmBlk, 0, stream>>>(
            abf, D, wo_t + (size_t)l * D * D, D, h, D, bo + (size_t)l * D, M, D, D);
        layernorm_kernel<<<M, 256, 0, stream>>>(h, ln2_g + (size_t)l * D, ln2_b + (size_t)l * D,
                                                abf, Ff, D);
        dim3 gDF(DFF / TN, M / TM);
        gemm_bf16_wmma_kernel<true, 0><<<gDF, gemmBlk, 0, stream>>>(
            abf, D, w1_t + (size_t)l * D * DFF, D, T1, DFF, b1 + (size_t)l * DFF, M, DFF, D);
        {
            int n = M * DFF;
            gelu_cast_kernel<<<(n + 255) / 256, 256, 0, stream>>>(T1, t1bf, n);
        }
        gemm_bf16_wmma_kernel<true, 2><<<gDD, gemmBlk, 0, stream>>>(
            t1bf, DFF, w2_t + (size_t)l * DFF * D, DFF, h, D, b2 + (size_t)l * D, M, D, DFF);
        add1_kernel<<<addBlocks, 256, 0, stream>>>(h, Ff, nMD);
    }

    layernorm_kernel<<<M, 256, 0, stream>>>(h, lnf_g, lnf_b, abf, (float*)nullptr, D);
    dim3 gDV(V / TN, M / TM);
    gemm_bf16_wmma_kernel<true, 0><<<gDV, gemmBlk, 0, stream>>>(
        abf, D, wout_t, D, logits, V, b_out, M, V, D);
    zero_kernel<<<1, 32, 0, stream>>>(lossp, 1);
    loss_kernel<<<M, 256, 0, stream>>>(logits, targets, lossp, V, 1.0f / (float)M);
}
